// MLastVectorAttention_79611513799196
// MI455X (gfx1250) — matrix operand order repaired and checked by reading, not compiled, not run on hardware
//
#include <hip/hip_runtime.h>
#include <hip/hip_bf16.h>
#include <math.h>
#include <stdint.h>

// Problem constants (match reference)
#define B_    4096
#define S_    10
#define E_    1536
#define H_    12
#define HD_   128
#define BS_   (B_ * S_)
#define SCALE_ 0.08838834764831845f   // 128^-0.5
#define LN_EPS_ 1e-5f

typedef __bf16 bf16;
typedef __attribute__((ext_vector_type(16))) __bf16 v16bf;
typedef __attribute__((ext_vector_type(8)))  float  v8f;
typedef unsigned int u32x4 __attribute__((ext_vector_type(4)));

union Frag32B {             // 32-byte WMMA operand fragment
    u32x4 u[2];
    v16bf v;
};

// ---------------------------------------------------------------------------
// CDNA5 async global->LDS staging.
// Toolchain probe (round-2 diagnostic) showed the builtin's params are
// pointers to GCC-vector `int __attribute__((vector_size(16)))` with
// global / LDS address spaces -> cast to exactly those types.
// ---------------------------------------------------------------------------
typedef int v4i __attribute__((vector_size(16)));
typedef v4i __attribute__((address_space(1))) v4i_g;   // global (AS1)
typedef v4i __attribute__((address_space(3))) v4i_l;   // LDS    (AS3)

#if __has_builtin(__builtin_amdgcn_global_load_async_to_lds_b128)
#define HAVE_ASYNC_LDS 1
#define ASYNC_COPY_B128(g, l)                                                  \
    __builtin_amdgcn_global_load_async_to_lds_b128(                            \
        (v4i_g*)(uintptr_t)(g),                                                \
        (v4i_l*)(unsigned)(uintptr_t)(l),                                      \
        0, 0)
#else
#define HAVE_ASYNC_LDS 0
#define ASYNC_COPY_B128(g, l) (*(u32x4*)(l) = *(const u32x4*)(g))
#endif

#if HAVE_ASYNC_LDS
#if __has_builtin(__builtin_amdgcn_s_wait_asynccnt)
#define WAIT_ASYNCCNT(n) __builtin_amdgcn_s_wait_asynccnt(n)
#else
#define WAIT_ASYNCCNT(n) asm volatile("s_wait_asynccnt %0" ::"i"(n) : "memory")
#endif
#else
#define WAIT_ASYNCCNT(n) ((void)0)
#endif

// ---------------------------------------------------------------------------
// Kernel 1: f32 [K][N] weight -> bf16 [N][K] transposed weight (LDS-tiled).
// One-time cost; makes the GEMM B-tile per-lane contiguous so both A and B
// tiles stage via GLOBAL_LOAD_ASYNC_TO_LDS_B128.
// ---------------------------------------------------------------------------
__global__ __launch_bounds__(256)
void cvt_transpose_kernel(const float* __restrict__ W, bf16* __restrict__ Wt) {
    __shared__ float t[32][33];
    const int bk = blockIdx.x * 32;        // K base
    const int bn = blockIdx.y * 32;        // N base
    const int tx = threadIdx.x & 31;
    const int ty = threadIdx.x >> 5;       // 0..7
#pragma unroll
    for (int i = 0; i < 32; i += 8)
        t[ty + i][tx] = W[(size_t)(bk + ty + i) * E_ + (bn + tx)];
    __syncthreads();
#pragma unroll
    for (int i = 0; i < 32; i += 8)
        Wt[(size_t)(bn + ty + i) * E_ + (bk + tx)] = (bf16)t[tx][ty + i];
}

// ---------------------------------------------------------------------------
// Kernel 2: LayerNorm(x + PE) -> bf16 activations. One 256-thread block/row.
// ---------------------------------------------------------------------------
__global__ __launch_bounds__(256)
void ln_pe_kernel(const float* __restrict__ x,
                  const float* __restrict__ gamma,
                  const float* __restrict__ beta,
                  bf16* __restrict__ xn) {
    const int row  = blockIdx.x;          // 0..BS-1
    const int s    = row % S_;
    const int tid  = threadIdx.x;
    const int lane = tid & 31;
    const int wave = tid >> 5;
    const float kdiv = -9.210340371976184f / (float)E_;  // -ln(10000)/E

    float v[6];
    float sum = 0.f, ssq = 0.f;
#pragma unroll
    for (int i = 0; i < 6; ++i) {
        int   c   = tid + i * 256;
        float div = __expf((float)(c & ~1) * kdiv);
        float ang = (float)s * div;
        float pe  = (c & 1) ? __cosf(ang) : __sinf(ang);
        float t   = x[(size_t)row * E_ + c] + pe;
        v[i] = t;
        sum += t;
        ssq += t * t;
    }
#pragma unroll
    for (int off = 16; off; off >>= 1) {
        sum += __shfl_xor(sum, off, 32);
        ssq += __shfl_xor(ssq, off, 32);
    }
    __shared__ float rsum[8], rssq[8];
    if (lane == 0) { rsum[wave] = sum; rssq[wave] = ssq; }
    __syncthreads();
    float ts = 0.f, tq = 0.f;
#pragma unroll
    for (int w = 0; w < 8; ++w) { ts += rsum[w]; tq += rssq[w]; }
    const float mu  = ts * (1.f / (float)E_);
    const float var = tq * (1.f / (float)E_) - mu * mu;
    const float rs  = rsqrtf(var + LN_EPS_);
#pragma unroll
    for (int i = 0; i < 6; ++i) {
        int c = tid + i * 256;
        float o = (v[i] - mu) * rs * gamma[c] + beta[c];
        xn[(size_t)row * E_ + c] = (bf16)o;
    }
}

// ---------------------------------------------------------------------------
// Kernel 3: gather last valid row per batch (prefix-valid mask, True=pad)
// ---------------------------------------------------------------------------
__global__ __launch_bounds__(256)
void gather_last_kernel(const unsigned char* __restrict__ mask,
                        const bf16* __restrict__ xn,
                        bf16* __restrict__ last) {
    const int b = blockIdx.x;
    int len = 0;
#pragma unroll
    for (int s = 0; s < S_; ++s) len += (mask[b * S_ + s] == 0) ? 1 : 0;
    const int idx = len - 1;
    const bf16* src = xn + ((size_t)b * S_ + idx) * E_;
    bf16*       dst = last + (size_t)b * E_;
    for (int e = threadIdx.x; e < E_; e += 256) dst[e] = src[e];
}

// ---------------------------------------------------------------------------
// Kernel 4: bf16 WMMA GEMM  C[M,N=E] = A[M,K=E] * Wt[N,K]^T + bias (+gelu)
// BM=BN=128, BK=32; 256 threads = 8 wave32s in 2x4 grid; wave = 64x32 subtile
// = 4x2 v_wmma_f32_16x16x32_bf16 accumulators. Tiles staged to double-
// buffered LDS with GLOBAL_LOAD_ASYNC_TO_LDS_B128 (ASYNCcnt pipeline).
// mode 0: store bf16 ; mode 1: exact GELU, store f32
// ---------------------------------------------------------------------------
#define BM  128
#define BN  128
#define BK  32
#define KT  (E_ / BK)   // 48
#define LDT 40          // bf16 elems per padded LDS row (32 + 8 pad)

__global__ __launch_bounds__(256)
void gemm_bf16_kernel(const bf16* __restrict__ A,
                      const bf16* __restrict__ Wt,   // [N][K] bf16
                      const float* __restrict__ bias,
                      bf16* __restrict__ Cbf,
                      float* __restrict__ Cf,
                      int mode) {
    __shared__ __align__(16) bf16 As[2][BM * LDT];
    __shared__ __align__(16) bf16 Bs[2][BN * LDT];

    const int tid  = threadIdx.x;
    const int lane = tid & 31;
    const int wave = tid >> 5;
    const size_t m0 = (size_t)blockIdx.x * BM;
    const int    n0 = blockIdx.y * BN;

    const int wm = (wave >> 2) * 64;   // wave row base within tile
    const int wn = (wave & 3) * 32;    // wave col base within tile

    // -- cooperative tile staging: thread t owns 16 bf16 of each tile --------
    const int crow = tid >> 1;               // 0..127
    const int chal = (tid & 1) << 4;         // 0 / 16
    const bf16* aSrc = A  + (m0 + crow) * (size_t)E_ + chal;
    const bf16* bSrc = Wt + (size_t)(n0 + crow) * E_ + chal;
    const int   dOff = crow * LDT + chal;    // LDS element offset

    // -- LDS fragment offsets (ISA VGPR layouts) -----------------------------
    // A 16x32: lane holds row m=lane%16, K chunks [k0..k0+7],[k0+16..k0+23],
    //          k0 = 8*(lane/16)
    int aOff[4];
#pragma unroll
    for (int mi = 0; mi < 4; ++mi)
        aOff[mi] = (wm + mi * 16 + (lane & 15)) * LDT + ((lane >> 4) << 3);
    // B 32x16: lane holds col n=lane%16, K chunks [k0..k0+7],[k0+16..k0+23],
    //          k0 = 8*(lane/16)
    int bOff[2];
#pragma unroll
    for (int ni = 0; ni < 2; ++ni)
        bOff[ni] = (wn + ni * 16 + (lane & 15)) * LDT + ((lane >> 4) << 3);

    v8f acc[4][2];
    const v8f vzero = {0.f, 0.f, 0.f, 0.f, 0.f, 0.f, 0.f, 0.f};
#pragma unroll
    for (int mi = 0; mi < 4; ++mi)
#pragma unroll
        for (int ni = 0; ni < 2; ++ni) acc[mi][ni] = vzero;

    // stage k-tile kt into LDS buffer `buf` (4 async b128 per thread)
    auto stage = [&](int kt, int buf) {
        const bf16* ag = aSrc + kt * BK;
        const bf16* bg = bSrc + kt * BK;
        ASYNC_COPY_B128(ag,     &As[buf][dOff]);
        ASYNC_COPY_B128(ag + 8, &As[buf][dOff + 8]);
        ASYNC_COPY_B128(bg,     &Bs[buf][dOff]);
        ASYNC_COPY_B128(bg + 8, &Bs[buf][dOff + 8]);
    };

    stage(0, 0);                              // prologue fill
    for (int kt = 0; kt < KT; ++kt) {
        const int cur = kt & 1;
        if (kt + 1 < KT) {
            stage(kt + 1, cur ^ 1);           // overlap next fill with compute
            WAIT_ASYNCCNT(4);                 // tile kt landed (in-order)
        } else {
            WAIT_ASYNCCNT(0);
        }
        if (kt + 2 < KT) {                    // L2 prefetch two tiles ahead
            __builtin_prefetch(aSrc + (kt + 2) * BK, 0, 1);
            __builtin_prefetch(bSrc + (kt + 2) * BK, 0, 1);
        }
        __syncthreads();                      // all waves' tile-kt data visible

        const bf16* Ab = As[cur];
        const bf16* Bb = Bs[cur];
        Frag32B bfr[2];
#pragma unroll
        for (int ni = 0; ni < 2; ++ni) {
            bfr[ni].u[0] = *(const u32x4*)(Bb + bOff[ni]);
            bfr[ni].u[1] = *(const u32x4*)(Bb + bOff[ni] + 16);
        }
#pragma unroll
        for (int mi = 0; mi < 4; ++mi) {
            Frag32B afr;
            afr.u[0] = *(const u32x4*)(Ab + aOff[mi]);
            afr.u[1] = *(const u32x4*)(Ab + aOff[mi] + 16);
#pragma unroll
            for (int ni = 0; ni < 2; ++ni)
                acc[mi][ni] = __builtin_amdgcn_wmma_f32_16x16x32_bf16(
                    false, afr.v, false, bfr[ni].v,
                    (short)0, acc[mi][ni], false, false);
        }
        __syncthreads();                      // reads done -> buffer reusable
    }

    // epilogue: C/D layout -> lanes 0-15: M=r, N=lane ; lanes 16-31: M=r+8
    const int gc0 = n0 + wn + (lane & 15);
    const int gr0 = (int)m0 + wm + ((lane >> 4) << 3);
#pragma unroll
    for (int mi = 0; mi < 4; ++mi) {
#pragma unroll
        for (int ni = 0; ni < 2; ++ni) {
            const int   gc = gc0 + ni * 16;
            const float bb = bias[gc];
#pragma unroll
            for (int r = 0; r < 8; ++r) {
                const size_t row = (size_t)(gr0 + mi * 16 + r);
                float val = acc[mi][ni][r] + bb;
                if (mode == 1) {
                    val = 0.5f * val * (1.0f + erff(val * 0.7071067811865476f));
                    Cf[row * E_ + gc] = val;
                } else {
                    Cbf[row * E_ + gc] = (bf16)val;
                }
            }
        }
    }
}

// ---------------------------------------------------------------------------
// Kernel 5: single-query attention per (b,h). 128 threads = HD lanes.
// ---------------------------------------------------------------------------
__global__ __launch_bounds__(128)
void attn_kernel(const bf16* __restrict__ Q,
                 const bf16* __restrict__ K,
                 const bf16* __restrict__ V,
                 const unsigned char* __restrict__ mask,
                 bf16* __restrict__ ctx) {
    const int bh = blockIdx.x;
    const int b  = bh / H_;
    const int h  = bh % H_;
    const int d  = threadIdx.x;
    const int lane = d & 31;
    const int wave = d >> 5;

    const float qd = (float)Q[(size_t)b * E_ + h * HD_ + d];

    float part[S_];
#pragma unroll
    for (int s = 0; s < S_; ++s) {
        float kv = (float)K[((size_t)b * S_ + s) * E_ + h * HD_ + d];
        part[s] = qd * kv;
    }
#pragma unroll
    for (int s = 0; s < S_; ++s)
#pragma unroll
        for (int off = 16; off; off >>= 1)
            part[s] += __shfl_xor(part[s], off, 32);

    __shared__ float red[S_][4];
    if (lane == 0)
#pragma unroll
        for (int s = 0; s < S_; ++s) red[s][wave] = part[s];
    __syncthreads();

    float sc[S_];
    float mx = -1e30f;
#pragma unroll
    for (int s = 0; s < S_; ++s) {
        float t = (red[s][0] + red[s][1] + red[s][2] + red[s][3]) * SCALE_;
        if (mask[b * S_ + s]) t = -1e9f;
        sc[s] = t;
        mx = fmaxf(mx, t);
    }
    float den = 0.f;
#pragma unroll
    for (int s = 0; s < S_; ++s) { sc[s] = __expf(sc[s] - mx); den += sc[s]; }
    const float inv = 1.f / den;

    float acc = 0.f;
#pragma unroll
    for (int s = 0; s < S_; ++s) {
        float vv = (float)V[((size_t)b * S_ + s) * E_ + h * HD_ + d];
        acc += sc[s] * inv * vv;
    }
    ctx[(size_t)b * E_ + h * HD_ + d] = (bf16)acc;
}

// ---------------------------------------------------------------------------
// Host orchestration
// ---------------------------------------------------------------------------
extern "C" void kernel_launch(void* const* d_in, const int* in_sizes, int n_in,
                              void* d_out, int out_size, void* d_ws, size_t ws_size,
                              hipStream_t stream) {
    (void)in_sizes; (void)n_in; (void)out_size; (void)ws_size;

    const float* a_vec = (const float*)d_in[0];
    const float* b_vec = (const float*)d_in[1];
    const unsigned char* mask_a = (const unsigned char*)d_in[2];
    const unsigned char* mask_b = (const unsigned char*)d_in[3];
    const float* qw = (const float*)d_in[4];  const float* qb = (const float*)d_in[5];
    const float* kw = (const float*)d_in[6];  const float* kb = (const float*)d_in[7];
    const float* vw = (const float*)d_in[8];  const float* vb = (const float*)d_in[9];
    const float* ow = (const float*)d_in[10]; const float* ob = (const float*)d_in[11];
    const float* fw = (const float*)d_in[12]; const float* fb = (const float*)d_in[13];
    const float* lng = (const float*)d_in[14];
    const float* lnb = (const float*)d_in[15];

    // workspace carve-up
    char* ws  = (char*)d_ws;
    size_t off = 0;
    auto carve = [&](size_t bytes) -> char* {
        char* p = ws + off;
        off += (bytes + 255) & ~(size_t)255;
        return p;
    };
    const size_t EE = (size_t)E_ * E_;
    bf16* wqT = (bf16*)carve(EE * sizeof(bf16));
    bf16* wkT = (bf16*)carve(EE * sizeof(bf16));
    bf16* wvT = (bf16*)carve(EE * sizeof(bf16));
    bf16* woT = (bf16*)carve(EE * sizeof(bf16));
    bf16* wfT = (bf16*)carve(EE * sizeof(bf16));
    bf16* xn   = (bf16*)carve((size_t)BS_ * E_ * sizeof(bf16));
    bf16* Kact = (bf16*)carve((size_t)BS_ * E_ * sizeof(bf16));
    bf16* Vact = (bf16*)carve((size_t)BS_ * E_ * sizeof(bf16));
    bf16* last = (bf16*)carve((size_t)B_ * E_ * sizeof(bf16));
    bf16* Qact = (bf16*)carve((size_t)B_ * E_ * sizeof(bf16));
    bf16* ctx  = (bf16*)carve((size_t)B_ * E_ * sizeof(bf16));
    bf16* out1 = (bf16*)carve((size_t)B_ * E_ * sizeof(bf16));

    // convert + transpose weights once: f32 [K][N] -> bf16 [N][K]
    {
        dim3 blk(256), grd(E_ / 32, E_ / 32);
        cvt_transpose_kernel<<<grd, blk, 0, stream>>>(qw, wqT);
        cvt_transpose_kernel<<<grd, blk, 0, stream>>>(kw, wkT);
        cvt_transpose_kernel<<<grd, blk, 0, stream>>>(vw, wvT);
        cvt_transpose_kernel<<<grd, blk, 0, stream>>>(ow, woT);
        cvt_transpose_kernel<<<grd, blk, 0, stream>>>(fw, wfT);
    }

    const dim3 gBig(BS_ / BM, E_ / BN);   // 320 x 12
    const dim3 gSml(B_  / BM, E_ / BN);   // 32  x 12

    for (int br = 0; br < 2; ++br) {
        const float*         xin = br ? b_vec  : a_vec;
        const unsigned char* msk = br ? mask_b : mask_a;
        float* outp = (float*)d_out + (size_t)br * B_ * E_;

        ln_pe_kernel<<<BS_, 256, 0, stream>>>(xin, lng, lnb, xn);
        gather_last_kernel<<<B_, 256, 0, stream>>>(msk, xn, last);

        gemm_bf16_kernel<<<gBig, 256, 0, stream>>>(xn, wkT, kb, Kact, nullptr, 0);
        gemm_bf16_kernel<<<gBig, 256, 0, stream>>>(xn, wvT, vb, Vact, nullptr, 0);
        gemm_bf16_kernel<<<gSml, 256, 0, stream>>>(last, wqT, qb, Qact, nullptr, 0);

        attn_kernel<<<B_ * H_, 128, 0, stream>>>(Qact, Kact, Vact, msk, ctx);

        gemm_bf16_kernel<<<gSml, 256, 0, stream>>>(ctx, woT, ob, out1, nullptr, 0);
        gemm_bf16_kernel<<<gSml, 256, 0, stream>>>(out1, wfT, fb, nullptr, outp, 1);
    }
}
